// Rank2SymmetricTensorHead_35416300323174
// MI455X (gfx1250) — compile-verified, matrix operand order checked by reading, not run on hardware
//
#include <hip/hip_runtime.h>
#include <hip/hip_bf16.h>
#include <stdint.h>

// ---------------------------------------------------------------------------
// Rank2SymmetricTensorHead for MI455X (gfx1250, wave32).
// CSR-regroup edges by destination node (node_emb read once per node, no float
// atomics for the segment mean), fuse sph-harmonics + LN + outer-product
// accumulation per node, then (N*9,128)@(128,128) GEMM via
// v_wmma_f32_16x16x32_f16 fused with SiLU, W2 and the batch segment-mean.
// This round: vectorized LDS fragment paths (ds_load_b128 / ds_store_b64)
// via a transposed W1 LDS layout with bank-spread padding.
// ---------------------------------------------------------------------------

#define NN 8192
#define EE 131072
#define CC 128
#define BB 64
#define MTOT (NN * 9)          // 73728 GEMM rows, divisible by 16
#define SA 136                 // A-tile LDS row stride (halves): 272B, 16B-aligned, bank spread 4
#define SW 136                 // W1t LDS row stride (halves)

typedef __attribute__((ext_vector_type(16))) _Float16 v16h;
typedef __attribute__((ext_vector_type(8)))  _Float16 v8h;
typedef __attribute__((ext_vector_type(4)))  _Float16 v4h;
typedef __attribute__((ext_vector_type(8)))  float    v8f;
typedef __attribute__((ext_vector_type(4)))  float    v4f;

// ---------------- CSR build ----------------

__global__ void k_hist(const int* __restrict__ dst, int* __restrict__ counts) {
    int e = blockIdx.x * blockDim.x + threadIdx.x;
    if (e < EE) atomicAdd(&counts[dst[e]], 1);
}

__global__ void k_bcnt(const int* __restrict__ batch, int* __restrict__ bcnt) {
    int n = blockIdx.x * blockDim.x + threadIdx.x;
    if (n < NN) atomicAdd(&bcnt[batch[n]], 1);
}

// exclusive scan of 8192 counts with one 256-thread block (32 items/thread)
__global__ void k_scan(const int* __restrict__ counts, int* __restrict__ offsets) {
    __shared__ int part[256];
    int tid = threadIdx.x;
    int local[32];
    int s = 0;
#pragma unroll
    for (int i = 0; i < 32; ++i) { local[i] = s; s += counts[tid * 32 + i]; }
    part[tid] = s;
    __syncthreads();
    if (tid == 0) {
        int run = 0;
        for (int i = 0; i < 256; ++i) { int t = part[i]; part[i] = run; run += t; }
    }
    __syncthreads();
    int base = part[tid];
#pragma unroll
    for (int i = 0; i < 32; ++i) offsets[tid * 32 + i] = base + local[i];
    if (tid == 255) offsets[NN] = base + s;   // == EE
}

__global__ void k_scatter(const int* __restrict__ dst, const int* __restrict__ offsets,
                          int* __restrict__ cursor, int* __restrict__ perm) {
    int e = blockIdx.x * blockDim.x + threadIdx.x;
    if (e < EE) {
        int d = dst[e];
        int p = atomicAdd(&cursor[d], 1);
        perm[offsets[d] + p] = e;
    }
}

// ---------------- fused per-node edge kernel ----------------
// One 128-thread block per node. Thread t owns channel c = t.
// Produces nt[n][k][c] = mean_e( LN(sum_l sh[e,l]*emb[l,c]) * op[e,k] ).

__global__ void __launch_bounds__(128)
k_edge(const float* __restrict__ node_emb, const float* __restrict__ evec,
       const int* __restrict__ offsets, const int* __restrict__ perm,
       const float* __restrict__ sph_w, const float* __restrict__ ln_w,
       const float* __restrict__ ln_b, float* __restrict__ nt) {
    const int n = blockIdx.x;
    const int c = threadIdx.x;
    const int lane = c & 31, wave = c >> 5;
    __shared__ float redA[4], redB[4];

    float emb[9];
    const float* ep = node_emb + (size_t)n * 9 * CC + c;
#pragma unroll
    for (int l = 0; l < 9; ++l) emb[l] = ep[l * CC];

    const float w0 = sph_w[0], w1 = sph_w[1], w2 = sph_w[2];
    const float lw = ln_w[c], lb = ln_b[c];
    const float sh0 = w0 * rsqrtf(1.0f + 1e-6f);
    const float s3 = 1.7320508075688772f, s15 = 3.872983346207417f, s5 = 2.23606797749979f;

    float acc[9];
#pragma unroll
    for (int k = 0; k < 9; ++k) acc[k] = 0.0f;

    const int e0 = offsets[n], e1 = offsets[n + 1];
    for (int i = e0; i < e1; ++i) {
        const int e = perm[i];
        const float vx = evec[3 * e + 0], vy = evec[3 * e + 1], vz = evec[3 * e + 2];
        const float r2 = vx * vx + vy * vy + vz * vz;

        // sph_l012 + per-block RMS norm
        float sh[9];
        sh[0] = sh0;
        const float inv1 = w1 * rsqrtf(r2 + 1e-6f);          // mean((s3*v)^2)=r2
        sh[1] = s3 * vx * inv1; sh[2] = s3 * vy * inv1; sh[3] = s3 * vz * inv1;
        const float t0 = s15 * vx * vy, t1 = s15 * vy * vz;
        const float t2 = 0.5f * s5 * (3.0f * vz * vz - r2);
        const float t3 = s15 * vx * vz;
        const float t4 = 0.5f * s15 * (vx * vx - vy * vy);
        const float m2 = (t0 * t0 + t1 * t1 + t2 * t2 + t3 * t3 + t4 * t4) * 0.2f;
        const float inv2 = w2 * rsqrtf(m2 + 1e-6f);
        sh[4] = t0 * inv2; sh[5] = t1 * inv2; sh[6] = t2 * inv2;
        sh[7] = t3 * inv2; sh[8] = t4 * inv2;

        float y = 0.0f;
#pragma unroll
        for (int l = 0; l < 9; ++l) y += sh[l] * emb[l];

        // LayerNorm over the 128 channels (wave shuffle + LDS cross-wave)
        float s = y, q = y * y;
#pragma unroll
        for (int off = 16; off > 0; off >>= 1) {
            s += __shfl_xor(s, off, 32);
            q += __shfl_xor(q, off, 32);
        }
        if (lane == 0) { redA[wave] = s; redB[wave] = q; }
        __syncthreads();
        const float S = redA[0] + redA[1] + redA[2] + redA[3];
        const float Q = redB[0] + redB[1] + redB[2] + redB[3];
        __syncthreads();
        const float mu = S * (1.0f / 128.0f);
        const float var = Q * (1.0f / 128.0f) - mu * mu;
        const float xh = (y - mu) * rsqrtf(var + 1e-5f) * lw + lb;

        const float op[9] = { vx * vx, vx * vy, vx * vz,
                              vy * vx, vy * vy, vy * vz,
                              vz * vx, vz * vy, vz * vz };
#pragma unroll
        for (int k = 0; k < 9; ++k) acc[k] += xh * op[k];
    }

    const int cnt = e1 - e0;
    const float scale = 1.0f / (float)(cnt > 0 ? cnt : 1);
    float* outp = nt + (size_t)n * 9 * CC;
#pragma unroll
    for (int k = 0; k < 9; ++k) outp[k * CC + c] = acc[k] * scale;  // coalesced per k
}

// ---------------- WMMA GEMM + SiLU + W2 + batch reduction ----------------
// Rows = nt flattened (73728 x 128). Each wave owns one 16-row M tile and
// sweeps 8 N-tiles x 4 K-chunks with v_wmma_f32_16x16x32_f16.
// W1 is staged TRANSPOSED in LDS so each lane's B fragment is two contiguous
// 16B runs -> ds_load_b128; A fragments likewise.

__global__ void __launch_bounds__(128)
k_gemm(const float* __restrict__ nt, const float* __restrict__ W1,
       const float* __restrict__ b1, const float* __restrict__ W2,
       const float* __restrict__ b2, const int* __restrict__ batch,
       float* __restrict__ r2sum) {
    __shared__ _Float16 W1t[CC * SW];         // transposed: W1t[n*SW + k] = W1[k][n]
    __shared__ _Float16 As[4][16 * SA];       // 4 waves x padded 16x128 tile

    const int tid  = threadIdx.x;
    const int lane = tid & 31;
    const int wave = tid >> 5;
    const int rowbase = (blockIdx.x * 4 + wave) * 16;

    // Stage W1 transposed as f16 (coalesced global b32 reads)
    for (int i = tid; i < CC * CC; i += 128) {
        int k = i >> 7, n = i & 127;
        W1t[n * SW + k] = (_Float16)W1[i];
    }

    // Stage this wave's 16x128 A tile: float4 global loads -> packed v4h LDS stores
    {
        const v4f* src4 = (const v4f*)(nt + (size_t)rowbase * CC);
#pragma unroll 4
        for (int i = 0; i < 16; ++i) {
            int idx = i * 32 + lane;           // 0..511 float4s, contiguous
            v4f d = src4[idx];
            int m  = idx >> 5;                 // 32 float4 per 128-col row
            int k4 = (idx & 31) * 4;
            v4h h = { (_Float16)d[0], (_Float16)d[1], (_Float16)d[2], (_Float16)d[3] };
            *(v4h*)(&As[wave][m * SA + k4]) = h;   // 8B-aligned ds_store_b64
        }
    }
    __syncthreads();

    const int mrow  = lane & 15;               // A row / C col index
    const int kbase = (lane >> 4) * 8;         // K half-select per ISA layout

    // Hoist A fragments for the 4 K-chunks (same for every N tile)
    v16h afr[4];
#pragma unroll
    for (int kk = 0; kk < 4; ++kk) {
        const _Float16* ar = &As[wave][mrow * SA];
        const int k0 = kk * 32 + kbase;
        v8h lo = *(const v8h*)(ar + k0);        // 16B-aligned ds_load_b128
        v8h hi = *(const v8h*)(ar + k0 + 16);
        afr[kk] = __builtin_shufflevector(lo, hi, 0,1,2,3,4,5,6,7,8,9,10,11,12,13,14,15);
    }

    float partial[8];
#pragma unroll
    for (int r = 0; r < 8; ++r) partial[r] = 0.0f;

#pragma unroll
    for (int ntile = 0; ntile < 8; ++ntile) {
        const int ncol = ntile * 16 + mrow;
        const float b1n = b1[ncol];
        const float w2n = W2[ncol];
        const _Float16* br = &W1t[ncol * SW];

        v8f cacc;
#pragma unroll
        for (int r = 0; r < 8; ++r) cacc[r] = b1n;   // + b1 via C init

#pragma unroll
        for (int kk = 0; kk < 4; ++kk) {
            const int k0 = kk * 32 + kbase;
            v8h lo = *(const v8h*)(br + k0);    // 16B-aligned ds_load_b128
            v8h hi = *(const v8h*)(br + k0 + 16);
            v16h b = __builtin_shufflevector(lo, hi, 0,1,2,3,4,5,6,7,8,9,10,11,12,13,14,15);
            cacc = __builtin_amdgcn_wmma_f32_16x16x32_f16(
                false, afr[kk], false, b, (short)0, cacc, false, false);
        }

        // SiLU then dot with W2 column (accumulate per row across N)
#pragma unroll
        for (int r = 0; r < 8; ++r) {
            const float h = cacc[r];
            const float sv = h / (1.0f + __expf(-h));
            partial[r] += sv * w2n;
        }
    }

    // Reduce over the 16 lanes of each half-wave (cols of this row group)
#pragma unroll
    for (int off = 8; off >= 1; off >>= 1) {
#pragma unroll
        for (int r = 0; r < 8; ++r) partial[r] += __shfl_xor(partial[r], off, 16);
    }

    if ((lane & 15) == 0) {
        const int mhalf = (lane >> 4) * 8;     // lanes16-31 hold rows M=r+8
        const float b2v = b2[0];
#pragma unroll
        for (int r = 0; r < 8; ++r) {
            const int row = rowbase + mhalf + r;      // global GEMM row = n*9+l
            const int nnode = row / 9;
            const int l = row - nnode * 9;
            const float val = partial[r] + b2v;
            atomicAdd(&r2sum[batch[nnode] * 9 + l], val);
        }
    }
}

__global__ void k_final(const float* __restrict__ r2sum, const int* __restrict__ bcnt,
                        float* __restrict__ out) {
    int i = threadIdx.x;
    if (i < BB * 9) {
        int b = i / 9;
        int cnt = bcnt[b];
        out[i] = r2sum[i] / (float)(cnt > 0 ? cnt : 1);
    }
}

// ---------------- launcher ----------------

extern "C" void kernel_launch(void* const* d_in, const int* in_sizes, int n_in,
                              void* d_out, int out_size, void* d_ws, size_t ws_size,
                              hipStream_t stream) {
    const float* node_emb = (const float*)d_in[0];            // (N,9,C)
    const float* evec     = (const float*)d_in[1];            // (E,3)
    const int*   eidx     = (const int*)d_in[2];              // (2,E)
    const int*   batch    = (const int*)d_in[3];              // (N,)
    const float* sph_w    = (const float*)d_in[4];            // (3,)
    const float* ln_w     = (const float*)d_in[5];            // (C,)
    const float* ln_b     = (const float*)d_in[6];            // (C,)
    const float* W1       = (const float*)d_in[7];            // (C,C)
    const float* b1       = (const float*)d_in[8];            // (C,)
    const float* W2       = (const float*)d_in[9];            // (C,1)
    const float* b2       = (const float*)d_in[10];           // (1,)
    float* out = (float*)d_out;                               // (B,9)

    const int* dst = eidx + EE;                               // edge_index[1]

    // workspace layout
    int*   counts  = (int*)d_ws;                              // N
    int*   cursor  = counts + NN;                             // N
    float* r2sum   = (float*)(cursor + NN);                   // B*9
    int*   bcnt    = (int*)(r2sum + BB * 9);                  // B
    int*   offsets = bcnt + BB;                               // N+1
    int*   perm    = offsets + (NN + 1);                      // E
    // align nt to 256B so float4 loads in k_gemm are legal
    float* nt      = (float*)(((uintptr_t)(perm + EE) + 255) & ~(uintptr_t)255);

    const size_t zero_bytes = (size_t)(NN + NN + BB * 9 + BB) * 4;
    hipMemsetAsync(d_ws, 0, zero_bytes, stream);

    k_hist   <<<EE / 256, 256, 0, stream>>>(dst, counts);
    k_bcnt   <<<NN / 256, 256, 0, stream>>>(batch, bcnt);
    k_scan   <<<1, 256, 0, stream>>>(counts, offsets);
    k_scatter<<<EE / 256, 256, 0, stream>>>(dst, offsets, cursor, perm);
    k_edge   <<<NN, 128, 0, stream>>>(node_emb, evec, offsets, perm,
                                      sph_w, ln_w, ln_b, nt);
    k_gemm   <<<MTOT / 64, 128, 0, stream>>>(nt, W1, b1, W2, b2, batch, r2sum);
    k_final  <<<1, BB * 9, 0, stream>>>(r2sum, bcnt, out);
}
